// MultiHeadAttention_9345848836480
// MI455X (gfx1250) — matrix operand order repaired and checked by reading, not compiled, not run on hardware
//
#include <hip/hip_runtime.h>
#include <math.h>

// ---------------------------------------------------------------------------
// MHA for B=1, S=4096, E=1024, H=16, D=64 on gfx1250 (wave32, WMMA).
//   cvt      : x -> f16 ; W* -> f16 transposed [N][K]   (one-time, cheap)
//   gemm<QK> : Q = x*Wq+bq  -> f16 [H][S][D]
//   gemm<QK> : K = x*Wk+bk  -> f16 [H][S][D]
//   gemm<VT> : V = x*Wv+bv  -> f16 [H][D][S]  (transposed for P*V B-operand)
//   attn     : flash-style streaming softmax, WMMA for QK^T and P*V
//   gemm<F32>: out = O*Wo+bo -> fp32 d_out
// GEMM: 128x128x64 block, double-buffered LDS, async global->LDS (ASYNCcnt),
// prefetch of tile k+2.  Attn: wave owns 32 queries, per-wave LDS softmax.
// ---------------------------------------------------------------------------

typedef _Float16 half_t;
typedef __attribute__((ext_vector_type(16))) _Float16 v16h;
typedef __attribute__((ext_vector_type(8)))  float    v8f;

constexpr int SEQ = 4096;
constexpr int EMB = 1024;
constexpr int NH  = 16;
constexpr int HD  = 64;

constexpr int BLK_M = 128;
constexpr int BLK_N = 128;
constexpr int BLK_K = 64;
constexpr int LDA_S = BLK_K + 8;  // 72 halves -> 144B row stride (16B aligned)
constexpr int LDB_S = BLK_K + 8;  // Bs stored [N][K], stride 72 halves

union FragH { v16h v; uint4 q[2]; };
union AccF  { v8f  v; float f[8]; };

enum { MODE_F32 = 0, MODE_QK = 1, MODE_VT = 2 };

// gfx1250 async global->LDS (ASYNCcnt) if the toolchain exposes it.
#if __has_builtin(__builtin_amdgcn_global_load_async_to_lds_b128) && \
    __has_builtin(__builtin_amdgcn_s_wait_asynccnt)
#define HAVE_ASYNC_LDS 1
typedef int v4i_ __attribute__((vector_size(16)));
#define AS_GLOBAL __attribute__((address_space(1)))
#define AS_LDS    __attribute__((address_space(3)))
#else
#define HAVE_ASYNC_LDS 0
#endif

__device__ __forceinline__ void lds_copy_b128(const half_t* src, half_t* dst) {
#if HAVE_ASYNC_LDS
  __builtin_amdgcn_global_load_async_to_lds_b128(
      (AS_GLOBAL v4i_*)src, (AS_LDS v4i_*)dst, 0, 0);
#else
  *(uint4*)dst = *(const uint4*)src;
#endif
}

__device__ __forceinline__ void async_wait() {
#if HAVE_ASYNC_LDS
  __builtin_amdgcn_s_wait_asynccnt(0);
#endif
}

// Per-wave LDS ordering: LDS ops from one wave complete in order (ISA 7.1);
// we only need to stop the compiler from reordering across the fence.
__device__ __forceinline__ void wave_fence() { asm volatile("" ::: "memory"); }

// ---------------------------------------------------------------------------
// One-time conversions.
// ---------------------------------------------------------------------------
__global__ __launch_bounds__(256) void cvt_f16_kernel(
    const float* __restrict__ in, half_t* __restrict__ out) {
  int i = (blockIdx.x * 256 + threadIdx.x) * 4;
  float4 d = *(const float4*)(in + i);
  half_t* p = out + i;
  p[0] = (half_t)d.x; p[1] = (half_t)d.y;
  p[2] = (half_t)d.z; p[3] = (half_t)d.w;
}

// Wt[n][k] = (f16) W[k][n]  -- LDS-tiled 32x32 transpose
__global__ __launch_bounds__(256) void cvt_t_kernel(
    const float* __restrict__ W, half_t* __restrict__ Wt) {
  __shared__ half_t tile[32][33];
  const int bk = blockIdx.x * 32;
  const int bn = blockIdx.y * 32;
  const int tx = threadIdx.x & 31;
  const int ty = threadIdx.x >> 5;   // 0..7
#pragma unroll
  for (int j = 0; j < 4; j++) {
    int k = ty + j * 8;
    tile[k][tx] = (half_t)W[(size_t)(bk + k) * EMB + bn + tx];
  }
  __syncthreads();
#pragma unroll
  for (int j = 0; j < 4; j++) {
    int n = ty + j * 8;
    Wt[(size_t)(bn + n) * EMB + bk + tx] = tile[tx][n];
  }
}

// ---------------------------------------------------------------------------
// Tiled GEMM: C[4096 x 1024] = A[4096 x 1024] * Wt^T + bias   (A, Wt in f16,
// Wt pre-transposed to [N][K]).  256 threads = 8 waves (4 M x 2 N);
// wave tile 32x64 = 2x4 WMMA tiles, 2 k-substeps per 64-wide K tile.
// ---------------------------------------------------------------------------
template <int MODE>
__global__ __launch_bounds__(256) void gemm_kernel(
    const half_t* __restrict__ A, const half_t* __restrict__ Bt,
    const float* __restrict__ bias, void* __restrict__ Out)
{
  __shared__ half_t As[2][BLK_M * LDA_S];   // [row][k]
  __shared__ half_t Bs[2][BLK_N * LDB_S];   // [n][k]

  const int tid    = threadIdx.x;
  const int lane   = tid & 31;
  const int wave   = tid >> 5;      // 0..7
  const int wm     = wave >> 1;     // 0..3
  const int wn     = wave & 1;      // 0..1
  const int m0     = blockIdx.x * BLK_M;
  const int n0     = blockIdx.y * BLK_N;
  const int half16 = lane >> 4;     // 0 / 1
  const int l16    = lane & 15;

  auto load_tiles = [&](int buf, int k0) {
    // A tile 128 x 64 halves and B tile 128 x 64 halves: 1024 b128 chunks each
#pragma unroll
    for (int i = 0; i < 4; i++) {
      int id = tid + i * 256;
      int r  = id >> 3;          // 0..127
      int c8 = id & 7;           // 8-half chunk
      lds_copy_b128(A + (size_t)(m0 + r) * EMB + k0 + c8 * 8,
                    &As[buf][r * LDA_S + c8 * 8]);
    }
#pragma unroll
    for (int i = 0; i < 4; i++) {
      int id = tid + i * 256;
      int r  = id >> 3;          // n row 0..127
      int c8 = id & 7;
      lds_copy_b128(Bt + (size_t)(n0 + r) * EMB + k0 + c8 * 8,
                    &Bs[buf][r * LDB_S + c8 * 8]);
    }
  };

  AccF acc[2][4];
#pragma unroll
  for (int i = 0; i < 2; i++)
#pragma unroll
    for (int j = 0; j < 4; j++)
#pragma unroll
      for (int e = 0; e < 8; e++) acc[i][j].f[e] = 0.0f;

  load_tiles(0, 0);
  async_wait();
  __syncthreads();

  for (int k0 = 0; k0 < EMB; k0 += BLK_K) {
    const int cur = (k0 >> 6) & 1;

    // issue loads for tile k+1 into the alternate buffer (overlaps compute)
    if (k0 + BLK_K < EMB) load_tiles(cur ^ 1, k0 + BLK_K);

    // prefetch tile k+2 into near caches (global_prefetch_b8)
    if (k0 + 2 * BLK_K < EMB) {
      const int kp  = k0 + 2 * BLK_K;
      const int idx = tid & 127;
      __builtin_prefetch(A  + (size_t)(m0 + idx) * EMB + kp, 0, 3);
      __builtin_prefetch(Bt + (size_t)(n0 + idx) * EMB + kp, 0, 3);
    }

    // ---- two 32-wide k-substeps from the current buffer ----
#pragma unroll
    for (int ks = 0; ks < 2; ks++) {
      FragH a[2], b[4];
#pragma unroll
      for (int mi = 0; mi < 2; mi++) {
        int row = wm * 32 + mi * 16 + l16;
        // A layout: lanes0-15: K 0..7,16..23 ; lanes16-31: K 8..15,24..31
        const half_t* p = &As[cur][row * LDA_S + ks * 32 + half16 * 8];
        a[mi].q[0] = *(const uint4*)p;
        a[mi].q[1] = *(const uint4*)(p + 16);
      }
#pragma unroll
      for (int ni = 0; ni < 4; ni++) {
        int col = wn * 64 + ni * 16 + l16;
        // B layout: lane = N, K 0..7,16..23 (lanes0-15) ; 8..15,24..31 (lanes16-31)
        const uint4* p =
            (const uint4*)&Bs[cur][col * LDB_S + ks * 32 + half16 * 8];
        b[ni].q[0] = p[0];
        b[ni].q[1] = p[2];
      }
#pragma unroll
      for (int mi = 0; mi < 2; mi++)
#pragma unroll
        for (int ni = 0; ni < 4; ni++)
          acc[mi][ni].v = __builtin_amdgcn_wmma_f32_16x16x32_f16(
              false, a[mi].v, false, b[ni].v, (short)0, acc[mi][ni].v,
              false, false);
    }

    async_wait();      // next-tile async copies landed in LDS
    __syncthreads();   // and everyone is done reading the current buffer
  }

  // ---- epilogue: bias + store per mode ----
#pragma unroll
  for (int mi = 0; mi < 2; mi++) {
#pragma unroll
    for (int ni = 0; ni < 4; ni++) {
      int col = n0 + wn * 64 + ni * 16 + l16;
      float bv = bias[col];
#pragma unroll
      for (int i = 0; i < 8; i++) {
        int row = m0 + wm * 32 + mi * 16 + i + half16 * 8;
        float v = acc[mi][ni].f[i] + bv;
        if (MODE == MODE_F32) {
          ((float*)Out)[(size_t)row * EMB + col] = v;
        } else if (MODE == MODE_QK) {
          int h = col >> 6, d = col & 63;
          ((half_t*)Out)[(size_t)h * SEQ * HD + (size_t)row * HD + d] = (half_t)v;
        } else {  // MODE_VT: V transposed [H][D][S]
          int h = col >> 6, d = col & 63;
          ((half_t*)Out)[(size_t)h * HD * SEQ + (size_t)d * SEQ + row] = (half_t)v;
        }
      }
    }
  }
}

// ---------------------------------------------------------------------------
// Flash attention: 8 waves/block, each wave owns 32 queries x D=64 for head h.
// Per 32-key block: QK^T (8 WMMA) -> online softmax via per-wave LDS (wave
// fences only, no block barriers) -> P*V (8 WMMA) with rescaled accumulators.
// ---------------------------------------------------------------------------
__global__ __launch_bounds__(256) void attn_kernel(
    const half_t* __restrict__ Qh, const half_t* __restrict__ Kh,
    const half_t* __restrict__ Vt, half_t* __restrict__ O)
{
  constexpr int WAVES = 8;
  __shared__ float  Sc[WAVES][32][33];   // scores, padded
  __shared__ half_t Pb[WAVES][32][32];   // probabilities (f16), A-frag layout
  __shared__ float  Al[WAVES][32];       // per-row rescale alpha
  __shared__ float  Ls[WAVES][32];       // per-row 1/l

  const int tid    = threadIdx.x;
  const int lane   = tid & 31;
  const int wave   = tid >> 5;
  const int h      = blockIdx.y;
  const int q0     = (blockIdx.x * WAVES + wave) * 32;
  const int half16 = lane >> 4;
  const int l16    = lane & 15;

  const half_t* Qbase = Qh + (size_t)h * SEQ * HD;
  const half_t* Kbase = Kh + (size_t)h * SEQ * HD;
  const half_t* Vbase = Vt + (size_t)h * HD * SEQ;

  // Q fragments held in registers: aq[mi][ks], ks=0 -> d 0..31, ks=1 -> 32..63
  FragH aq[2][2];
#pragma unroll
  for (int mi = 0; mi < 2; mi++)
#pragma unroll
    for (int ks = 0; ks < 2; ks++) {
      const half_t* p =
          Qbase + (size_t)(q0 + mi * 16 + l16) * HD + ks * 32 + half16 * 8;
      aq[mi][ks].q[0] = *(const uint4*)p;
      aq[mi][ks].q[1] = *(const uint4*)(p + 16);
    }

  AccF o[2][4];
#pragma unroll
  for (int mi = 0; mi < 2; mi++)
#pragma unroll
    for (int nt = 0; nt < 4; nt++)
#pragma unroll
      for (int e = 0; e < 8; e++) o[mi][nt].f[e] = 0.0f;

  float m_r = -1e30f, l_r = 0.0f;   // lane owns query row q0+lane
  const float scale = 0.125f;       // 1/sqrt(64)

  for (int kb = 0; kb < SEQ / 32; kb++) {
    const int kbase = kb * 32;

    // prefetch next key/value block into near caches
    if (kb + 1 < SEQ / 32) {
      __builtin_prefetch(Kbase + (size_t)(kbase + 32 + lane) * HD, 0, 3);
      __builtin_prefetch(Vbase + (size_t)lane * SEQ + kbase + 32, 0, 3);
      __builtin_prefetch(Vbase + (size_t)(lane + 32) * SEQ + kbase + 32, 0, 3);
    }

    // K^T fragments: B operand, N = key index, K = d
    FragH bk[2][2];
#pragma unroll
    for (int nt = 0; nt < 2; nt++) {
      const half_t* rowp = Kbase + (size_t)(kbase + nt * 16 + l16) * HD;
#pragma unroll
      for (int ks = 0; ks < 2; ks++) {
        const half_t* p = rowp + ks * 32 + half16 * 8;
        bk[nt][ks].q[0] = *(const uint4*)p;
        bk[nt][ks].q[1] = *(const uint4*)(p + 16);
      }
    }
    // V fragments: independent of softmax -> issue loads early so they
    // overlap the exp/max VALU work (fences below would otherwise pin them).
    FragH bv[4];
#pragma unroll
    for (int nt = 0; nt < 4; nt++) {
      const half_t* p =
          Vbase + (size_t)(nt * 16 + l16) * SEQ + kbase + half16 * 8;
      bv[nt].q[0] = *(const uint4*)p;
      bv[nt].q[1] = *(const uint4*)(p + 16);
    }

    // scores S = Q * K^T  (32 x 32)
#pragma unroll
    for (int mi = 0; mi < 2; mi++)
#pragma unroll
      for (int nt = 0; nt < 2; nt++) {
        AccF s;
#pragma unroll
        for (int e = 0; e < 8; e++) s.f[e] = 0.0f;
        s.v = __builtin_amdgcn_wmma_f32_16x16x32_f16(
            false, aq[mi][0].v, false, bk[nt][0].v, (short)0, s.v, false, false);
        s.v = __builtin_amdgcn_wmma_f32_16x16x32_f16(
            false, aq[mi][1].v, false, bk[nt][1].v, (short)0, s.v, false, false);
        // stage scores (C layout: VGPR i -> row, lane%16 -> col)
#pragma unroll
        for (int i = 0; i < 8; i++)
          Sc[wave][mi * 16 + i + half16 * 8][nt * 16 + l16] = s.f[i];
      }
    wave_fence();

    // online softmax: lane owns query row r = lane (two LDS passes)
    {
      float mx = m_r;
#pragma unroll
      for (int j = 0; j < 32; j++)
        mx = fmaxf(mx, Sc[wave][lane][j] * scale);
      float alpha = __expf(m_r - mx);
      float suml = 0.0f;
#pragma unroll
      for (int j = 0; j < 32; j++) {
        float p = __expf(Sc[wave][lane][j] * scale - mx);
        suml += p;
        Pb[wave][lane][j] = (half_t)p;
      }
      l_r = l_r * alpha + suml;
      m_r = mx;
      Al[wave][lane] = alpha;
    }
    wave_fence();

    // rescale output accumulators by alpha[row]
#pragma unroll
    for (int mi = 0; mi < 2; mi++)
#pragma unroll
      for (int e = 0; e < 8; e++) {
        float alpha = Al[wave][mi * 16 + e + half16 * 8];
#pragma unroll
        for (int nt = 0; nt < 4; nt++) o[mi][nt].f[e] *= alpha;
      }

    // P fragments (A operand, 16 x 32 over keys)
    FragH ap[2];
#pragma unroll
    for (int mi = 0; mi < 2; mi++) {
      const half_t* p = &Pb[wave][mi * 16 + l16][half16 * 8];
      ap[mi].q[0] = *(const uint4*)p;
      ap[mi].q[1] = *(const uint4*)(p + 16);
    }
#pragma unroll
    for (int mi = 0; mi < 2; mi++)
#pragma unroll
      for (int nt = 0; nt < 4; nt++)
        o[mi][nt].v = __builtin_amdgcn_wmma_f32_16x16x32_f16(
            false, ap[mi].v, false, bv[nt].v, (short)0, o[mi][nt].v, false, false);
  }

  // finalize: divide by l and write O as f16 [S][E] row-major
  Ls[wave][lane] = 1.0f / l_r;
  wave_fence();
#pragma unroll
  for (int mi = 0; mi < 2; mi++)
#pragma unroll
    for (int e = 0; e < 8; e++) {
      float inv = Ls[wave][mi * 16 + e + half16 * 8];
      int row = q0 + mi * 16 + e + half16 * 8;
#pragma unroll
      for (int nt = 0; nt < 4; nt++)
        O[(size_t)row * EMB + h * HD + nt * 16 + l16] =
            (half_t)(o[mi][nt].f[e] * inv);
    }
}

// ---------------------------------------------------------------------------
extern "C" void kernel_launch(void* const* d_in, const int* in_sizes, int n_in,
                              void* d_out, int out_size, void* d_ws, size_t ws_size,
                              hipStream_t stream)
{
  const float* x  = (const float*)d_in[0];
  const float* Wq = (const float*)d_in[1];
  const float* bq = (const float*)d_in[2];
  const float* Wk = (const float*)d_in[3];
  const float* bk = (const float*)d_in[4];
  const float* Wv = (const float*)d_in[5];
  const float* bv = (const float*)d_in[6];
  const float* Wo = (const float*)d_in[7];
  const float* bo = (const float*)d_in[8];

  char* ws = (char*)d_ws;
  const size_t MB = (size_t)1024 * 1024;
  half_t* xh  = (half_t*)(ws + 0 * MB);    // [S][E]    f16  (8 MB)
  half_t* Wtq = (half_t*)(ws + 8 * MB);    // [N][K]    f16  (2 MB each)
  half_t* Wtk = (half_t*)(ws + 10 * MB);
  half_t* Wtv = (half_t*)(ws + 12 * MB);
  half_t* Wto = (half_t*)(ws + 14 * MB);
  half_t* Qh  = (half_t*)(ws + 16 * MB);   // [H][S][D] f16  (8 MB)
  half_t* Kh  = (half_t*)(ws + 24 * MB);   // [H][S][D] f16
  half_t* Vt  = (half_t*)(ws + 32 * MB);   // [H][D][S] f16
  half_t* Oo  = (half_t*)(ws + 40 * MB);   // [S][E]    f16

  // one-time conversions
  cvt_f16_kernel<<<SEQ * EMB / (256 * 4), 256, 0, stream>>>(x, xh);
  dim3 tgrid(EMB / 32, EMB / 32);
  cvt_t_kernel<<<tgrid, 256, 0, stream>>>(Wq, Wtq);
  cvt_t_kernel<<<tgrid, 256, 0, stream>>>(Wk, Wtk);
  cvt_t_kernel<<<tgrid, 256, 0, stream>>>(Wv, Wtv);
  cvt_t_kernel<<<tgrid, 256, 0, stream>>>(Wo, Wto);

  dim3 ggrid(SEQ / BLK_M, EMB / BLK_N);    // 32 x 8 blocks

  gemm_kernel<MODE_QK><<<ggrid, 256, 0, stream>>>(xh, Wtq, bq, Qh);
  gemm_kernel<MODE_QK><<<ggrid, 256, 0, stream>>>(xh, Wtk, bk, Kh);
  gemm_kernel<MODE_VT><<<ggrid, 256, 0, stream>>>(xh, Wtv, bv, Vt);

  attn_kernel<<<dim3(SEQ / (32 * 8), NH), 256, 0, stream>>>(Qh, Kh, Vt, Oo);

  gemm_kernel<MODE_F32><<<ggrid, 256, 0, stream>>>(Oo, Wto, bo, d_out);
}
